// GNNModel_51153060496038
// MI455X (gfx1250) — matrix operand order repaired and checked by reading, not compiled, not run on hardware
//
#include <hip/hip_runtime.h>
#include <cstdint>
#include <cstddef>

// Problem constants (match reference)
#define NN 20000           // nodes
#define NE 320000          // raw edges
#define NT (NE + NN)       // edges including self-loops
#define NH 5               // attention heads

typedef __attribute__((ext_vector_type(16))) _Float16 v16h;
typedef __attribute__((ext_vector_type(8)))  float    v8f;
typedef __attribute__((ext_vector_type(4)))  unsigned int v4u;
typedef __attribute__((ext_vector_type(8)))  int      v8i;
typedef __attribute__((ext_vector_type(4)))  int      v4i;

#if defined(__HIP_DEVICE_COMPILE__) && __has_builtin(__builtin_amdgcn_tensor_load_to_lds)
#define HAVE_TDM 1
#else
#define HAVE_TDM 0
#endif

typedef __attribute__((address_space(3))) float lds_float;

__device__ __forceinline__ float softplus_f(float x) {
  return x > 20.0f ? x : log1pf(expf(x));
}
__device__ __forceinline__ float leaky_f(float x, float s) {
  return x > 0.0f ? x : s * x;
}
// src/dst for edge e; e >= NE are the appended self-loops
__device__ __forceinline__ void edge_sd(const int* ei, int e, int& s, int& d) {
  if (e < NE) { s = ei[e]; d = ei[NE + e]; }
  else        { s = e - NE; d = s; }
}

// ---------------------------------------------------------------------------
// WMMA GEMM: C[NN,Cout] = act(A[NN,K] @ W[K,Cout] + bias)
// One wave per block; each wave owns MT=5 16-row tiles x one 16-col tile, so
// the B fragment is reused 5x from registers. K is a template param -> fully
// unrolled K loop (K/32 WMMAs per tile). The 32x16 f32 weight tile for each
// k-step is staged into LDS by the Tensor Data Mover (D# built per ISA §8.3/8.4),
// waited with s_wait_tensorcnt, and read back as the B fragment via ds loads.
// act: 0 = none, 1 = softplus.
// ---------------------------------------------------------------------------
template <int K>
__global__ __launch_bounds__(32)
void k_wmma_gemm(const float* __restrict__ A, const float* __restrict__ W,
                 const float* __restrict__ bias, float* __restrict__ C,
                 int Cout, int act)
{
  constexpr int MT = 5;                 // 16-row tiles per wave (20000 = 250*80)
  __shared__ float lds_w[32 * 16];      // one 32x16 f32 weight tile

  const int lane    = threadIdx.x;
  const int hsel    = lane >> 4;        // half-wave select
  const int l16     = lane & 15;
  const int row0    = blockIdx.x * (16 * MT);
  const int colbase = blockIdx.y * 16;
  const int col     = colbase + l16;

  v8f acc[MT];
  const float bv = bias ? bias[col] : 0.0f;   // uniform branch
#pragma unroll
  for (int t = 0; t < MT; ++t)
#pragma unroll
    for (int i = 0; i < 8; ++i) acc[t][i] = bv;

#if HAVE_TDM
  const unsigned lds_base = (unsigned)(size_t)(lds_float*)lds_w;  // LDS byte offset
#endif

#pragma unroll
  for (int kt = 0; kt < K; kt += 32) {
    // ---- stage B tile: W[kt..kt+31, colbase..colbase+15] -> lds_w (row-major) ----
#if HAVE_TDM
    {
      const unsigned long long ga =
          (unsigned long long)(const void*)(W + (size_t)kt * Cout + colbase);
      // D# group 0: count=1 (user desc), lds_addr, 57-bit global addr, type=2
      const v4u g0 = {1u,
                      lds_base,
                      (unsigned)(ga & 0xffffffffu),
                      (unsigned)((ga >> 32) & 0x01ffffffu) | (2u << 30)};
      // D# group 1: data_size=4B; tensor_dim0=Cout, tensor_dim1=32;
      // tile_dim0=16 (contig), tile_dim1=32; tensor_dim0_stride=Cout
      const unsigned cu = (unsigned)Cout;
      const v8i g1 = {(int)(2u << 16),                       // data_size=2 (4 bytes)
                      (int)((cu & 0xffffu) << 16),           // tensor_dim0[15:0]
                      (int)(((cu >> 16) & 0xffffu) | (32u << 16)), // dim0 hi | dim1 lo
                      (int)(16u << 16),                      // dim1 hi=0 | tile_dim0=16
                      (int)(32u),                            // tile_dim1=32, tile_dim2=0
                      (int)cu,                               // tensor_dim0_stride[31:0]
                      0, 0};                                 // stride hi, dim1_stride
      const v4i gz4 = {0, 0, 0, 0};
      const v8i gz8 = {0, 0, 0, 0, 0, 0, 0, 0};
      // clang-23 / amdgpu-toolchain 6-arg form:
      // (uint32x4 g0, int32x8 g1, int32x4, int32x4, int32x8, i32 cpol)
      __builtin_amdgcn_tensor_load_to_lds(g0, g1, gz4, gz4, gz8, 0);
      __builtin_amdgcn_s_wait_tensorcnt(0);
      __asm__ volatile("" ::: "memory");   // keep ds reads after the TDM wait
    }
#else
    // fallback: cooperative staged load (coalesced 16-lane rows)
#pragma unroll
    for (int i = 0; i < 16; ++i) {
      const int idx = lane + 32 * i;       // 0..511 over 32x16 tile
      const int r = idx >> 4, c = idx & 15;
      lds_w[idx] = W[(size_t)(kt + r) * Cout + colbase + c];
    }
#endif

    // B 32x16 f16 fragment from LDS: lanes 0-15 hold K rows kt+0..15,
    // lanes 16-31 rows kt+16..31, column N = l16.
    v16h b;
#pragma unroll
    for (int j = 0; j < 16; ++j)
      b[j] = (_Float16)lds_w[(hsel * 16 + j) * 16 + l16];

    // 5 row tiles reuse this B fragment
#pragma unroll
    for (int t = 0; t < MT; ++t) {
      const float* ar = A + (size_t)(row0 + t * 16 + l16) * K;
      if (t == 0 && kt + 32 < K) __builtin_prefetch(ar + kt + 32, 0, 1);
      v16h a;
#pragma unroll
      for (int j = 0; j < 16; ++j) {
        a[j]     = (_Float16)ar[kt + hsel * 16 + j];
      }
      acc[t] = __builtin_amdgcn_wmma_f32_16x16x32_f16(
          false, a, false, b, (short)0, acc[t], false, false);
    }
  }

  // C/D f32 16x16 per tile: VGPR r holds row M = r + 8*hsel, col = l16
#pragma unroll
  for (int t = 0; t < MT; ++t) {
#pragma unroll
    for (int r = 0; r < 8; ++r) {
      const int orow = row0 + t * 16 + hsel * 8 + r;
      float v = acc[t][r];
      if (act == 1) v = softplus_f(v);
      C[(size_t)orow * Cout + col] = v;
    }
  }
}

// ---------------------------------------------------------------------------
// Elementwise / edge kernels
// ---------------------------------------------------------------------------
__global__ void k_fill(float* p, float v, int n) {
  int i = blockIdx.x * blockDim.x + threadIdx.x;
  if (i < n) p[i] = v;
}

__global__ void k_rowinit(float* out, const float* __restrict__ bias, int total, int C) {
  int i = blockIdx.x * blockDim.x + threadIdx.x;
  if (i < total) out[i] = bias[i % C];
}

__global__ void k_deg(const int* __restrict__ ei, float* deg) {
  int e = blockIdx.x * blockDim.x + threadIdx.x;
  if (e < NT) {
    int s, d; edge_sd(ei, e, s, d);
    atomicAdd(&deg[d], 1.0f);
  }
}

__global__ void k_dinv(float* deg) {
  int i = blockIdx.x * blockDim.x + threadIdx.x;
  if (i < NN) {
    float d = deg[i];
    deg[i] = d > 0.0f ? rsqrtf(d) : 0.0f;
  }
}

// GCN scatter (float4): out[dst,c:c+4] += h[src,c:c+4] * dinv[src]*dinv[dst]
__global__ void k_gcn_edge(const int* __restrict__ ei, const float* __restrict__ dinv,
                           const float* __restrict__ h, float* out, int C) {
  const int C4 = C >> 2;
  int idx = blockIdx.x * blockDim.x + threadIdx.x;
  int tot = NT * C4;
  if (idx < tot) {
    int e = idx / C4, c4 = idx % C4;
    int s, d; edge_sd(ei, e, s, d);
    const float norm = dinv[s] * dinv[d];
    const float4 hv = *(const float4*)(h + (size_t)s * C + c4 * 4);
    float* ob = out + (size_t)d * C + c4 * 4;
    atomicAdd(ob + 0, hv.x * norm);
    atomicAdd(ob + 1, hv.y * norm);
    atomicAdd(ob + 2, hv.z * norm);
    atomicAdd(ob + 3, hv.w * norm);
  }
}

// a_s[n,h] = <hx[n,h,:], att_src[h,:]>,  a_d likewise
__global__ void k_att_scores(const float* __restrict__ hx, const float* __restrict__ as,
                             const float* __restrict__ ad, float* sa, float* sd, int C) {
  int i = blockIdx.x * blockDim.x + threadIdx.x;
  if (i < NN * NH) {
    int n = i / NH, h = i % NH;
    const float* hv = hx + (size_t)n * NH * C + (size_t)h * C;
    float s0 = 0.0f, s1 = 0.0f;
    for (int c = 0; c < C; ++c) {
      float v = hv[c];
      s0 += v * as[h * C + c];
      s1 += v * ad[h * C + c];
    }
    sa[i] = s0;
    sd[i] = s1;
  }
}

// GAT pass 1: unnormalized softmax weights per (edge, head) + denominator scatter.
// exp without max-shift is mathematically identical (every node has a self-loop,
// scores are O(1e-2) at these weight scales).
__global__ void k_gat_e1(const int* __restrict__ ei, const float* __restrict__ sa,
                         const float* __restrict__ sd, float* wbuf, float* den) {
  int i = blockIdx.x * blockDim.x + threadIdx.x;
  if (i < NT * NH) {
    int e = i / NH, h = i % NH;
    int s, d; edge_sd(ei, e, s, d);
    float ev = leaky_f(sa[s * NH + h] + sd[d * NH + h], 0.2f);
    float w = expf(ev);
    wbuf[i] = w;
    atomicAdd(&den[d * NH + h], w);
  }
}

// GAT pass 2 (float4): out[dst,c:c+4] += (1/H) * sum_h alpha_{e,h} * hx[src,h,c:c+4]
__global__ void k_gat_e2(const int* __restrict__ ei, const float* __restrict__ hx,
                         const float* __restrict__ wbuf, const float* __restrict__ den,
                         float* out, int C) {
  const int C4 = C >> 2;
  int idx = blockIdx.x * blockDim.x + threadIdx.x;
  int tot = NT * C4;
  if (idx < tot) {
    int e = idx / C4, c4 = idx % C4;
    int s, d; edge_sd(ei, e, s, d);
    float ax = 0.0f, ay = 0.0f, az = 0.0f, aw = 0.0f;
#pragma unroll
    for (int h = 0; h < NH; ++h) {
      const float alpha = wbuf[e * NH + h] / den[d * NH + h];
      const float4 hv =
          *(const float4*)(hx + ((size_t)s * NH + h) * C + c4 * 4);
      ax += hv.x * alpha; ay += hv.y * alpha;
      az += hv.z * alpha; aw += hv.w * alpha;
    }
    const float sc = 1.0f / NH;
    float* ob = out + (size_t)d * C + c4 * 4;
    atomicAdd(ob + 0, ax * sc);
    atomicAdd(ob + 1, ay * sc);
    atomicAdd(ob + 2, az * sc);
    atomicAdd(ob + 3, aw * sc);
  }
}

// eval-mode BatchNorm1d + leaky_relu(0.01) + optional residual add
__global__ void k_bn_leaky(const float* __restrict__ xin, const float* __restrict__ g,
                           const float* __restrict__ b, const float* __restrict__ m,
                           const float* __restrict__ v, const float* __restrict__ resid,
                           float* out, int C) {
  int i = blockIdx.x * blockDim.x + threadIdx.x;
  if (i < NN * C) {
    int c = i % C;
    float val = (xin[i] - m[c]) * rsqrtf(v[c] + 1e-5f) * g[c] + b[c];
    val = leaky_f(val, 0.01f);
    if (resid) val += resid[i];
    out[i] = val;
  }
}

// xf = concat(skip1[N,128], x2[N,64]) -> [N,192]
__global__ void k_concat(const float* __restrict__ a, const float* __restrict__ b,
                         float* __restrict__ xf) {
  int i = blockIdx.x * blockDim.x + threadIdx.x;
  if (i < NN * 192) {
    int n = i / 192, c = i % 192;
    xf[i] = (c < 128) ? a[n * 128 + c] : b[n * 64 + (c - 128)];
  }
}

// scalar dense + softplus for small tail layers (dims not WMMA-tileable)
__global__ void k_dense_scalar(const float* __restrict__ A, const float* __restrict__ W,
                               const float* __restrict__ bias, float* out,
                               int Kin, int Kout) {
  int i = blockIdx.x * blockDim.x + threadIdx.x;
  if (i < NN * Kout) {
    int n = i / Kout, j = i % Kout;
    float s = bias[j];
    const float* ar = A + (size_t)n * Kin;
    for (int k = 0; k < Kin; ++k) s += ar[k] * W[k * Kout + j];
    out[i] = softplus_f(s);
  }
}

__global__ void k_mlp_final(const float* __restrict__ m3, const float* __restrict__ w4,
                            const float* __restrict__ b4, float* __restrict__ probs) {
  int n = blockIdx.x * blockDim.x + threadIdx.x;
  if (n < NN) {
    float s = b4[0];
    const float* ar = m3 + n * 24;
#pragma unroll
    for (int k = 0; k < 24; ++k) s += ar[k] * w4[k];
    probs[n] = 1.0f / (1.0f + expf(-s));
  }
}

// ---------------------------------------------------------------------------
// Host orchestration
// ---------------------------------------------------------------------------
#define GRID1(n) dim3((unsigned)(((n) + 255) / 256)), dim3(256)

static void launch_gemm(const float* A, const float* W, const float* bias, float* C,
                        int K, int Cout, int act, hipStream_t s) {
  dim3 grid(NN / 80, Cout / 16);   // 80 rows (5 tiles) per wave, 16 cols
  switch (K) {
    case 64:  k_wmma_gemm<64> <<<grid, 32, 0, s>>>(A, W, bias, C, Cout, act); break;
    case 128: k_wmma_gemm<128><<<grid, 32, 0, s>>>(A, W, bias, C, Cout, act); break;
    case 192: k_wmma_gemm<192><<<grid, 32, 0, s>>>(A, W, bias, C, Cout, act); break;
    default: break;  // unused
  }
}

extern "C" void kernel_launch(void* const* d_in, const int* in_sizes, int n_in,
                              void* d_out, int out_size, void* d_ws, size_t ws_size,
                              hipStream_t stream) {
  (void)in_sizes; (void)n_in; (void)out_size; (void)ws_size;

  // setup_inputs() dict order
  const float* x_in   = (const float*)d_in[0];
  const int*   ei     = (const int*)  d_in[1];   // [2, NE] flat (src row, dst row)
  const float* nn1_w1 = (const float*)d_in[2];
  const float* nn1_b1 = (const float*)d_in[3];
  const float* nn1_w2 = (const float*)d_in[4];
  const float* nn1_b2 = (const float*)d_in[5];
  const float* nn1_w3 = (const float*)d_in[6];
  const float* nn1_b3 = (const float*)d_in[7];
  const float* bn0_g = (const float*)d_in[8],  *bn0_b = (const float*)d_in[9];
  const float* bn0_m = (const float*)d_in[10], *bn0_v = (const float*)d_in[11];
  const float* bn1_g = (const float*)d_in[12], *bn1_b = (const float*)d_in[13];
  const float* bn1_m = (const float*)d_in[14], *bn1_v = (const float*)d_in[15];
  const float* bn2_g = (const float*)d_in[16], *bn2_b = (const float*)d_in[17];
  const float* bn2_m = (const float*)d_in[18], *bn2_v = (const float*)d_in[19];
  const float* gcn1_w = (const float*)d_in[20], *gcn1_b = (const float*)d_in[21];
  const float* gat1_w  = (const float*)d_in[22];
  const float* gat1_as = (const float*)d_in[23], *gat1_ad = (const float*)d_in[24];
  const float* gat1_b  = (const float*)d_in[25];
  const float* gcn2_w = (const float*)d_in[26], *gcn2_b = (const float*)d_in[27];
  const float* gat2_w  = (const float*)d_in[28];
  const float* gat2_as = (const float*)d_in[29], *gat2_ad = (const float*)d_in[30];
  const float* gat2_b  = (const float*)d_in[31];
  const float* mlp_w1 = (const float*)d_in[32], *mlp_b1 = (const float*)d_in[33];
  const float* mlp_w2 = (const float*)d_in[34], *mlp_b2 = (const float*)d_in[35];
  const float* mlp_w3 = (const float*)d_in[36], *mlp_b3 = (const float*)d_in[37];
  const float* mlp_w4 = (const float*)d_in[38], *mlp_b4 = (const float*)d_in[39];

  float* ws = (float*)d_ws;
  size_t off = 0;
  auto alloc = [&](size_t nf) { float* p = ws + off; off += nf; return p; };

  // ---- nn1 MLP: 64 -> 128 -> 64 -> 128, softplus/softplus/linear ----
  float* t1 = alloc((size_t)NN * 128);
  launch_gemm(x_in, nn1_w1, nn1_b1, t1, 64, 128, 1, stream);
  float* t2 = alloc((size_t)NN * 64);
  launch_gemm(t1, nn1_w2, nn1_b2, t2, 128, 64, 1, stream);
  float* t3 = alloc((size_t)NN * 128);
  launch_gemm(t2, nn1_w3, nn1_b3, t3, 64, 128, 0, stream);
  // BN0 + leaky -> x
  float* x = alloc((size_t)NN * 128);
  k_bn_leaky<<<GRID1(NN * 128), 0, stream>>>(t3, bn0_g, bn0_b, bn0_m, bn0_v, nullptr, x, 128);

  // ---- degrees / symmetric norm (shared by both GCN layers) ----
  float* dinv = alloc(NN);
  k_fill<<<GRID1(NN), 0, stream>>>(dinv, 0.0f, NN);
  k_deg<<<GRID1(NT), 0, stream>>>(ei, dinv);
  k_dinv<<<GRID1(NN), 0, stream>>>(dinv);

  // ---- GCN1: h1 = x @ W, aggregate ----
  float* h1 = alloc((size_t)NN * 128);
  launch_gemm(x, gcn1_w, nullptr, h1, 128, 128, 0, stream);
  float* x1a = alloc((size_t)NN * 128);
  k_rowinit<<<GRID1(NN * 128), 0, stream>>>(x1a, gcn1_b, NN * 128, 128);
  k_gcn_edge<<<GRID1(NT * 32), 0, stream>>>(ei, dinv, h1, x1a, 128);

  // ---- GAT1 (C=128, H=5) ----
  float* hx1 = alloc((size_t)NN * NH * 128);
  launch_gemm(x1a, gat1_w, nullptr, hx1, 128, NH * 128, 0, stream);
  float* sa1 = alloc((size_t)NN * NH);
  float* sd1 = alloc((size_t)NN * NH);
  k_att_scores<<<GRID1(NN * NH), 0, stream>>>(hx1, gat1_as, gat1_ad, sa1, sd1, 128);
  float* wb  = alloc((size_t)NT * NH);
  float* den = alloc((size_t)NN * NH);
  k_fill<<<GRID1(NN * NH), 0, stream>>>(den, 0.0f, NN * NH);
  k_gat_e1<<<GRID1(NT * NH), 0, stream>>>(ei, sa1, sd1, wb, den);
  float* x1b = alloc((size_t)NN * 128);
  k_rowinit<<<GRID1(NN * 128), 0, stream>>>(x1b, gat1_b, NN * 128, 128);
  k_gat_e2<<<GRID1(NT * 32), 0, stream>>>(ei, hx1, wb, den, x1b, 128);

  // ---- BN1 + leaky + skip: skip1 = x + leaky(bn1(x1b)) ----
  float* skip1 = alloc((size_t)NN * 128);
  k_bn_leaky<<<GRID1(NN * 128), 0, stream>>>(x1b, bn1_g, bn1_b, bn1_m, bn1_v, x, skip1, 128);

  // ---- GCN2: skip1[128] -> 64 ----
  float* h2 = alloc((size_t)NN * 64);
  launch_gemm(skip1, gcn2_w, nullptr, h2, 128, 64, 0, stream);
  float* x2a = alloc((size_t)NN * 64);
  k_rowinit<<<GRID1(NN * 64), 0, stream>>>(x2a, gcn2_b, NN * 64, 64);
  k_gcn_edge<<<GRID1(NT * 16), 0, stream>>>(ei, dinv, h2, x2a, 64);

  // ---- GAT2 (C=64, H=5) ----
  float* hx2 = alloc((size_t)NN * NH * 64);
  launch_gemm(x2a, gat2_w, nullptr, hx2, 64, NH * 64, 0, stream);
  float* sa2 = alloc((size_t)NN * NH);
  float* sd2 = alloc((size_t)NN * NH);
  k_att_scores<<<GRID1(NN * NH), 0, stream>>>(hx2, gat2_as, gat2_ad, sa2, sd2, 64);
  k_fill<<<GRID1(NN * NH), 0, stream>>>(den, 0.0f, NN * NH);
  k_gat_e1<<<GRID1(NT * NH), 0, stream>>>(ei, sa2, sd2, wb, den);
  float* x2b = alloc((size_t)NN * 64);
  k_rowinit<<<GRID1(NN * 64), 0, stream>>>(x2b, gat2_b, NN * 64, 64);
  k_gat_e2<<<GRID1(NT * 16), 0, stream>>>(ei, hx2, wb, den, x2b, 64);

  // ---- BN2 + leaky ----
  float* x2c = alloc((size_t)NN * 64);
  k_bn_leaky<<<GRID1(NN * 64), 0, stream>>>(x2b, bn2_g, bn2_b, bn2_m, bn2_v, nullptr, x2c, 64);

  // ---- concat -> d_out[:N*192], then head MLP ----
  float* xf = (float*)d_out;
  k_concat<<<GRID1(NN * 192), 0, stream>>>(skip1, x2c, xf);

  float* m1 = alloc((size_t)NN * 96);
  launch_gemm(xf, mlp_w1, mlp_b1, m1, 192, 96, 1, stream);   // 192 -> 96 (WMMA)
  float* m2 = alloc((size_t)NN * 48);
  k_dense_scalar<<<GRID1(NN * 48), 0, stream>>>(m1, mlp_w2, mlp_b2, m2, 96, 48);
  float* m3 = alloc((size_t)NN * 24);
  k_dense_scalar<<<GRID1(NN * 24), 0, stream>>>(m2, mlp_w3, mlp_b3, m3, 48, 24);
  k_mlp_final<<<GRID1(NN), 0, stream>>>(m3, mlp_w4, mlp_b4, xf + (size_t)NN * 192);
}
